// BaselineFeatureAct_45397804319426
// MI455X (gfx1250) — compile-verified, matrix operand order checked by reading, not run on hardware
//
#include <hip/hip_runtime.h>

// ---------------------------------------------------------------------------
// Types for CDNA5 WMMA (wave32): 16x16x32 bf16 -> f32
// ---------------------------------------------------------------------------
typedef __attribute__((ext_vector_type(16))) __bf16 v16bf;
typedef __attribute__((ext_vector_type(8)))  __bf16 v8bf;
typedef __attribute__((ext_vector_type(8)))  float  v8f;

#define EPSBN 1e-5f

// float -> bf16, round-to-nearest-even, via bit ops
__device__ __forceinline__ __bf16 f2bf(float f) {
    union { float f; unsigned u; } x; x.f = f;
    unsigned r = (x.u + 0x7FFFu + ((x.u >> 16) & 1u)) >> 16;
    unsigned short h = (unsigned short)r;
    return __builtin_bit_cast(__bf16, h);
}

// ---------------------------------------------------------------------------
// Prep kernels
// ---------------------------------------------------------------------------
__global__ void k_bnfold(const float* __restrict__ g, const float* __restrict__ bb,
                         const float* __restrict__ m, const float* __restrict__ v,
                         const float* __restrict__ cb,
                         float* __restrict__ scale, float* __restrict__ shift, int C) {
    int i = blockIdx.x * blockDim.x + threadIdx.x;
    if (i >= C) return;
    float inv = 1.0f, sh = 0.0f;
    if (g) { inv = g[i] * rsqrtf(v[i] + EPSBN); sh = bb[i] - m[i] * inv; }
    if (cb) sh += cb[i] * inv;
    scale[i] = inv; shift[i] = sh;
}

__global__ void k_cvt_w(const float* __restrict__ w, __bf16* __restrict__ wb,
                        int Co, int K, int Kp) {
    int i = blockIdx.x * blockDim.x + threadIdx.x;
    int n = Co * Kp; if (i >= n) return;
    int co = i / Kp, k = i - co * Kp;
    wb[i] = f2bf((k < K) ? w[(size_t)co * K + k] : 0.0f);
}

__global__ void k_cvt_a(const float* __restrict__ a, __bf16* __restrict__ ab, int n) {
    int i = blockIdx.x * blockDim.x + threadIdx.x;
    if (i < n) ab[i] = f2bf(a[i]);
}

// ---------------------------------------------------------------------------
// Implicit-GEMM conv with fused scale/shift (+ReLU).
// Block = 128 threads = 4 waves; each wave: COT x (16 Co) x (16 pos) tile.
// The im2col B fragment is gathered ONCE per K-chunk and reused across COT
// weight tiles (COT WMMAs per gather) -> amortizes the expensive gather.
// A fragments are two contiguous 16B loads matching the 16-bit A-matrix
// VGPR layout (lanes 0-15: K 0..7/16..23, lanes 16-31: K 8..15/24..31).
// ---------------------------------------------------------------------------
template <int KSZ, int COT>
__global__ __launch_bounds__(128) void k_conv_wmma(
    const float* __restrict__ x, const __bf16* __restrict__ wb,
    const float* __restrict__ scale, const float* __restrict__ shift,
    float* __restrict__ out, int Ci, int H, int W, int Co, int Kp, int relu)
{
    constexpr int KK  = KSZ * KSZ;
    constexpr int PAD = KSZ / 2;
    const int HW   = H * W;
    const int lane = threadIdx.x & 31;
    const int wave = threadIdx.x >> 5;
    const int b      = blockIdx.y;
    const int coBase = blockIdx.z * (16 * COT);
    const int posBase = blockIdx.x * 64 + wave * 16;
    const int kb   = (lane < 16) ? 0 : 8;
    const int l15  = lane & 15;
    const int pos  = posBase + l15;
    const bool posOK = (pos < HW);
    const int oh = pos / W;
    const int ow = pos - oh * W;
    const float*  xb   = x  + (size_t)b * Ci * HW;
    const __bf16* wrow = wb + (size_t)(coBase + l15) * Kp;

    v8f c[COT];
#pragma unroll
    for (int t = 0; t < COT; ++t) c[t] = (v8f){};

    for (int k0 = 0; k0 < Kp; k0 += 32) {
        // --- B fragment: im2col gather (once per chunk) ---
        v16bf Bf;
#pragma unroll
        for (int half = 0; half < 2; ++half) {
#pragma unroll
            for (int i = 0; i < 8; ++i) {
                int k  = k0 + half * 16 + kb + i;
                int ci = k / KK;
                int r  = k - ci * KK;
                int kh = r / KSZ;
                int kw = r - kh * KSZ;
                int iy = oh + kh - PAD;
                int ix = ow + kw - PAD;
                float v = 0.0f;
                if (posOK && ci < Ci &&
                    (unsigned)iy < (unsigned)H && (unsigned)ix < (unsigned)W)
                    v = xb[((size_t)ci * H + iy) * W + ix];
                Bf[half * 8 + i] = f2bf(v);
            }
        }
        // --- COT weight tiles share the gathered B fragment ---
#pragma unroll
        for (int t = 0; t < COT; ++t) {
            const __bf16* wr = wrow + (size_t)t * 16 * Kp;
            v8bf alo = *reinterpret_cast<const v8bf*>(wr + k0 + kb);
            v8bf ahi = *reinterpret_cast<const v8bf*>(wr + k0 + 16 + kb);
            v16bf A;
#pragma unroll
            for (int i = 0; i < 8; ++i) { A[i] = alo[i]; A[8 + i] = ahi[i]; }
            c[t] = __builtin_amdgcn_wmma_f32_16x16x32_bf16(false, A, false, Bf,
                                                           (short)0, c[t], false, false);
        }
    }
    if (posOK) {
        const int mAdd = (lane < 16) ? 0 : 8;   // C/D layout: lanes>=16 hold M=r+8
#pragma unroll
        for (int t = 0; t < COT; ++t) {
#pragma unroll
            for (int r = 0; r < 8; ++r) {
                int co = coBase + t * 16 + r + mAdd;
                float y = c[t][r] * scale[co] + shift[co];
                if (relu) y = fmaxf(y, 0.0f);
                out[((size_t)b * Co + co) * HW + pos] = y;
            }
        }
    }
}

// ---------------------------------------------------------------------------
// bf16 GEMM: D[M][N] = A[M][Kp] * B[N][Kp]^T with per-N scale/shift (+ReLU).
// Block = 128 threads (4 waves); wave does 16x16; grid = (N/16, M/64).
// ---------------------------------------------------------------------------
__global__ __launch_bounds__(128) void k_gemm_wmma(
    const __bf16* __restrict__ Ab, const __bf16* __restrict__ Bb,
    const float* __restrict__ scale, const float* __restrict__ shift,
    float* __restrict__ outF, __bf16* __restrict__ outB,
    int N, int Kp, int relu)
{
    const int lane = threadIdx.x & 31;
    const int wave = threadIdx.x >> 5;
    const int mBase = (blockIdx.y * 4 + wave) * 16;
    const int nBase = blockIdx.x * 16;
    const int kb  = (lane < 16) ? 0 : 8;
    const int l15 = lane & 15;
    const __bf16* arow = Ab + (size_t)(mBase + l15) * Kp;
    const __bf16* brow = Bb + (size_t)(nBase + l15) * Kp;

    v8f c = {};
    for (int k0 = 0; k0 < Kp; k0 += 32) {
        v8bf alo = *reinterpret_cast<const v8bf*>(arow + k0 + kb);
        v8bf ahi = *reinterpret_cast<const v8bf*>(arow + k0 + 16 + kb);
        v8bf blo = *reinterpret_cast<const v8bf*>(brow + k0 + kb);
        v8bf bhi = *reinterpret_cast<const v8bf*>(brow + k0 + 16 + kb);
        v16bf A, B;
#pragma unroll
        for (int i = 0; i < 8; ++i) {
            A[i] = alo[i]; A[8 + i] = ahi[i];
            B[i] = blo[i]; B[8 + i] = bhi[i];
        }
        c = __builtin_amdgcn_wmma_f32_16x16x32_bf16(false, A, false, B,
                                                    (short)0, c, false, false);
    }
    const int mAdd = (lane < 16) ? 0 : 8;
    const int n = nBase + l15;
#pragma unroll
    for (int r = 0; r < 8; ++r) {
        int m = mBase + r + mAdd;
        float y = c[r] * scale[n] + shift[n];
        if (relu) y = fmaxf(y, 0.0f);
        if (outF) outF[(size_t)m * N + n] = y;
        if (outB) outB[(size_t)m * N + n] = f2bf(y);
    }
}

// ---------------------------------------------------------------------------
// 2x2 max pool (stride 2), layout [B][C][H][W]
// ---------------------------------------------------------------------------
__global__ void k_pool(const float* __restrict__ in, float* __restrict__ out,
                       int C, int H, int W, int total) {
    int i = blockIdx.x * blockDim.x + threadIdx.x;
    if (i >= total) return;
    int OW = W >> 1, OH = H >> 1;
    int ow = i % OW;
    int t  = i / OW;
    int oh = t % OH;  t /= OH;
    int cc = t % C;
    int b  = t / C;
    const float* p = in + (((size_t)b * C + cc) * H + 2 * oh) * W + 2 * ow;
    out[i] = fmaxf(fmaxf(p[0], p[1]), fmaxf(p[W], p[W + 1]));
}

// s2[b][p] = sum_c f[b][c][p]^2
__global__ void k_sqsum(const float* __restrict__ f, float* __restrict__ s2,
                        int C, int HW, int total) {
    int i = blockIdx.x * blockDim.x + threadIdx.x;
    if (i >= total) return;
    int b = i / HW, p = i - b * HW;
    const float* fp = f + (size_t)b * C * HW + p;
    float s = 0.0f;
    for (int c = 0; c < C; ++c) { float v = fp[(size_t)c * HW]; s += v * v; }
    s2[i] = s;
}

// Deterministic two-pass dot reduction
__global__ void k_dotpart(const float* __restrict__ a, const float* __restrict__ b,
                          float* __restrict__ part, int n) {
    __shared__ float sm[256];
    float s = 0.0f;
    for (int i = blockIdx.x * 256 + threadIdx.x; i < n; i += gridDim.x * 256)
        s += a[i] * b[i];
    sm[threadIdx.x] = s; __syncthreads();
    for (int st = 128; st > 0; st >>= 1) {
        if ((int)threadIdx.x < st) sm[threadIdx.x] += sm[threadIdx.x + st];
        __syncthreads();
    }
    if (threadIdx.x == 0) part[blockIdx.x] = sm[0];
}

__global__ void k_reducefin(const float* __restrict__ part, float* __restrict__ out, int nb) {
    __shared__ float sm[256];
    sm[threadIdx.x] = ((int)threadIdx.x < nb) ? part[threadIdx.x] : 0.0f;
    __syncthreads();
    for (int st = 128; st > 0; st >>= 1) {
        if ((int)threadIdx.x < st) sm[threadIdx.x] += sm[threadIdx.x + st];
        __syncthreads();
    }
    if (threadIdx.x == 0) out[0] = sm[0];
}

// out = tgt_x * (1 + tgt_f * s2 / sqrt(normsum))
__global__ void k_featfin(const float* __restrict__ tgtx, const float* __restrict__ tgtf,
                          const float* __restrict__ s2, const float* __restrict__ normsum,
                          float* __restrict__ out, int C, int HW, int total) {
    int i = blockIdx.x * blockDim.x + threadIdx.x;
    if (i >= total) return;
    int b = i / (C * HW);
    int p = i % HW;
    float inv = rsqrtf(normsum[0]);
    out[i] = tgtx[i] * (1.0f + tgtf[i] * s2[(size_t)b * HW + p] * inv);
}

// ---------------------------------------------------------------------------
// Host orchestration
// ---------------------------------------------------------------------------
extern "C" void kernel_launch(void* const* d_in, const int* in_sizes, int n_in,
                              void* d_out, int out_size, void* d_ws, size_t ws_size,
                              hipStream_t stream) {
    (void)in_sizes; (void)n_in; (void)out_size; (void)ws_size;
    const int Bn = 512;

    auto F = [&](int i) { return (const float*)d_in[i]; };

    // ---- workspace bump allocator (256B aligned) ----
    size_t off = 0;
    auto alloc = [&](size_t bytes) -> void* {
        off = (off + 255) & ~(size_t)255;
        void* p = (char*)d_ws + off;
        off += bytes;
        return p;
    };
    auto allocF = [&](size_t n) { return (float*)alloc(n * sizeof(float)); };
    auto allocH = [&](size_t n) { return (__bf16*)alloc(n * sizeof(__bf16)); };

    float* bufA = allocF((size_t)Bn * 32 * 48 * 48);   // conv ping
    float* bufB = allocF((size_t)Bn * 32 * 48 * 48);   // conv pong
    float* x0   = allocF((size_t)Bn * 32 * 24 * 24);
    float* x1   = allocF((size_t)Bn * 64 * 12 * 12);
    float* x2   = allocF((size_t)Bn * 128 * 6 * 6);
    float* x3   = allocF((size_t)Bn * 256 * 3 * 3);
    float* xf01 = allocF((size_t)Bn * 64 * 12 * 12);
    float* xf12 = allocF((size_t)Bn * 128 * 6 * 6);
    float* xf23 = allocF((size_t)Bn * 256 * 3 * 3);
    float* srcf = allocF((size_t)Bn * 32 * 24 * 24);
    float* srcp = allocF((size_t)Bn * 32 * 12 * 12);
    float* tgtf = allocF((size_t)Bn * 64 * 12 * 12);
    float* s2b  = allocF((size_t)Bn * 144);
    float* t2b  = allocF((size_t)Bn * 144);
    float* partb = allocF(256);
    float* normb = allocF(4);

    struct CC { int wi, bi, bni, Ci, Co, H; };
    const CC cc[8] = {
        { 1,  2, 17,   1,  32, 48}, { 3,  4, 21,  32,  32, 48},
        { 5,  6, 25,  32,  64, 24}, { 7,  8, 29,  64,  64, 24},
        { 9, 10, 33,  64, 128, 12}, {11, 12, 37, 128, 128, 12},
        {13, 14, 41, 128, 256,  6}, {15, 16, 45, 256, 256,  6},
    };
    __bf16* cwB[8]; float* cSc[8]; float* cSh[8]; int cKp[8];
    for (int l = 0; l < 8; ++l) {
        int K = cc[l].Ci * 9;
        cKp[l] = (K + 31) & ~31;
        cwB[l] = allocH((size_t)cc[l].Co * cKp[l]);
        cSc[l] = allocF(cc[l].Co);
        cSh[l] = allocF(cc[l].Co);
    }
    struct FC1x1 { int wi, bi, C; };
    const FC1x1 fx[6] = {
        {49, 50,  32}, {51, 52,  64},   // fact0_1 src/tgt
        {53, 54,  64}, {55, 56, 128},   // fact1_2
        {57, 58, 128}, {59, 60, 256},   // fact2_3
    };
    __bf16* fwB[6]; float* fSc[6]; float* fSh[6];
    for (int l = 0; l < 6; ++l) {
        fwB[l] = allocH((size_t)fx[l].C * fx[l].C);
        fSc[l] = allocF(fx[l].C);
        fSh[l] = allocF(fx[l].C);
    }
    __bf16* fc1B = allocH((size_t)256 * 2304);
    __bf16* fc2B = allocH((size_t)128 * 256);
    float* fc1Sc = allocF(256); float* fc1Sh = allocF(256);
    float* fc2Sc = allocF(128); float* fc2Sh = allocF(128);
    __bf16* embB = allocH((size_t)Bn * 2304);
    __bf16* hB   = allocH((size_t)Bn * 256);

    // ---- prep: fold BN, convert weights ----
    for (int l = 0; l < 8; ++l) {
        int Co = cc[l].Co, K = cc[l].Ci * 9;
        hipLaunchKernelGGL(k_bnfold, dim3((Co + 63) / 64), dim3(64), 0, stream,
                           F(cc[l].bni), F(cc[l].bni + 1), F(cc[l].bni + 2), F(cc[l].bni + 3),
                           F(cc[l].bi), cSc[l], cSh[l], Co);
        int n = Co * cKp[l];
        hipLaunchKernelGGL(k_cvt_w, dim3((n + 255) / 256), dim3(256), 0, stream,
                           F(cc[l].wi), cwB[l], Co, K, cKp[l]);
    }
    for (int l = 0; l < 6; ++l) {
        int C = fx[l].C;
        hipLaunchKernelGGL(k_bnfold, dim3((C + 63) / 64), dim3(64), 0, stream,
                           (const float*)nullptr, (const float*)nullptr,
                           (const float*)nullptr, (const float*)nullptr,
                           F(fx[l].bi), fSc[l], fSh[l], C);
        int n = C * C;
        hipLaunchKernelGGL(k_cvt_w, dim3((n + 255) / 256), dim3(256), 0, stream,
                           F(fx[l].wi), fwB[l], C, C, C);
    }
    hipLaunchKernelGGL(k_bnfold, dim3(4), dim3(64), 0, stream,
                       F(63), F(64), F(65), F(66), F(62), fc1Sc, fc1Sh, 256);
    hipLaunchKernelGGL(k_bnfold, dim3(2), dim3(64), 0, stream,
                       F(69), F(70), F(71), F(72), F(68), fc2Sc, fc2Sh, 128);
    hipLaunchKernelGGL(k_cvt_w, dim3((256 * 2304 + 255) / 256), dim3(256), 0, stream,
                       F(61), fc1B, 256, 2304, 2304);
    hipLaunchKernelGGL(k_cvt_w, dim3((128 * 256 + 255) / 256), dim3(256), 0, stream,
                       F(67), fc2B, 128, 256, 256);

    // ---- generic conv launcher: COT=4 when Co%64==0 else COT=2 ----
    auto convK3 = [&](const float* in, int l, float* out) {
        int H = cc[l].H, HW = H * H, Co = cc[l].Co;
        if (Co % 64 == 0) {
            dim3 g((HW + 63) / 64, Bn, Co / 64);
            hipLaunchKernelGGL((k_conv_wmma<3, 4>), g, dim3(128), 0, stream,
                               in, cwB[l], cSc[l], cSh[l], out,
                               cc[l].Ci, H, H, Co, cKp[l], 1);
        } else {
            dim3 g((HW + 63) / 64, Bn, Co / 32);
            hipLaunchKernelGGL((k_conv_wmma<3, 2>), g, dim3(128), 0, stream,
                               in, cwB[l], cSc[l], cSh[l], out,
                               cc[l].Ci, H, H, Co, cKp[l], 1);
        }
    };
    auto convK1 = [&](const float* in, int fl, int H, float* out) {
        int C = fx[fl].C, HW = H * H;
        if (C % 64 == 0) {
            dim3 g((HW + 63) / 64, Bn, C / 64);
            hipLaunchKernelGGL((k_conv_wmma<1, 4>), g, dim3(128), 0, stream,
                               in, fwB[fl], fSc[fl], fSh[fl], out, C, H, H, C, C, 0);
        } else {
            dim3 g((HW + 63) / 64, Bn, C / 32);
            hipLaunchKernelGGL((k_conv_wmma<1, 2>), g, dim3(128), 0, stream,
                               in, fwB[fl], fSc[fl], fSh[fl], out, C, H, H, C, C, 0);
        }
    };
    auto pool = [&](const float* in, float* out, int C, int H) {
        int total = Bn * C * (H / 2) * (H / 2);
        hipLaunchKernelGGL(k_pool, dim3((total + 255) / 256), dim3(256), 0, stream,
                           in, out, C, H, H, total);
    };
    auto fact = [&](const float* srcx, int sl, int Hs,
                    const float* tgtx, int tl, int Ht, float* outx) {
        int Cs = fx[sl].C, Ct = fx[tl].C, HWt = Ht * Ht;
        convK1(srcx, sl, Hs, srcf);
        pool(srcf, srcp, Cs, Hs);
        int tot1 = Bn * HWt;
        hipLaunchKernelGGL(k_sqsum, dim3((tot1 + 255) / 256), dim3(256), 0, stream,
                           srcp, s2b, Cs, HWt, tot1);
        convK1(tgtx, tl, Ht, tgtf);
        hipLaunchKernelGGL(k_sqsum, dim3((tot1 + 255) / 256), dim3(256), 0, stream,
                           tgtf, t2b, Ct, HWt, tot1);
        hipLaunchKernelGGL(k_dotpart, dim3(128), dim3(256), 0, stream, s2b, t2b, partb, tot1);
        hipLaunchKernelGGL(k_reducefin, dim3(1), dim3(256), 0, stream, partb, normb, 128);
        int tot2 = Bn * Ct * HWt;
        hipLaunchKernelGGL(k_featfin, dim3((tot2 + 255) / 256), dim3(256), 0, stream,
                           tgtx, tgtf, s2b, normb, outx, Ct, HWt, tot2);
    };

    // ---- forward pipeline ----
    const float* X = F(0);
    convK3(X,    0, bufA);                // conv0  -> [B,32,48,48]
    convK3(bufA, 1, bufB);                // conv0b
    pool(bufB, x0, 32, 48);               // x0 [B,32,24,24]

    convK3(x0,   2, bufA);                // conv1  -> [B,64,24,24]
    convK3(bufA, 3, bufB);                // conv1b
    pool(bufB, x1, 64, 24);               // x1 [B,64,12,12]

    fact(x0, 0, 24, x1, 1, 12, xf01);     // fact0_1 -> [B,64,12,12]

    convK3(xf01, 4, bufA);                // conv2  -> [B,128,12,12]
    convK3(bufA, 5, bufB);                // conv2b
    pool(bufB, x2, 128, 12);              // x2 [B,128,6,6]

    fact(x1, 2, 12, x2, 3, 6, xf12);      // fact1_2 -> [B,128,6,6]

    convK3(xf12, 6, bufA);                // conv3  -> [B,256,6,6]
    convK3(bufA, 7, bufB);                // conv3b
    pool(bufB, x3, 256, 6);               // x3 [B,256,3,3]

    fact(x2, 4, 6, x3, 5, 3, xf23);       // fact2_3 -> [B,256,3,3] == emb

    // ---- FC head ----
    int nemb = Bn * 2304;
    hipLaunchKernelGGL(k_cvt_a, dim3((nemb + 255) / 256), dim3(256), 0, stream,
                       xf23, embB, nemb);
    hipLaunchKernelGGL(k_gemm_wmma, dim3(256 / 16, Bn / 64), dim3(128), 0, stream,
                       embB, fc1B, fc1Sc, fc1Sh, (float*)nullptr, hB, 256, 2304, 1);
    hipLaunchKernelGGL(k_gemm_wmma, dim3(128 / 16, Bn / 64), dim3(128), 0, stream,
                       hB, fc2B, fc2Sc, fc2Sh, (float*)d_out, (__bf16*)nullptr, 128, 256, 0);
}
